// SimpleRNN_12833362281151
// MI455X (gfx1250) — matrix operand order repaired and checked by reading, not compiled, not run on hardware
//
#include <hip/hip_runtime.h>
#include <hip/hip_bf16.h>
#include <math.h>

#define BB 64
#define TT 512
#define II 512
#define HH 1024
#define OO 512

typedef __bf16 bf16_t;
typedef __attribute__((ext_vector_type(16))) __bf16 v16bf;
typedef __attribute__((ext_vector_type(8)))  __bf16 v8bf;
typedef __attribute__((ext_vector_type(8)))  float  v8f;

union BF16x16 { v16bf v; v8bf h[2]; };

// One wave computes a 16x16 f32 tile of  A[MxK] * B[KxN]  where B is given
// pre-transposed as BT[NxK] (row-major). K must be a multiple of 32.
//
// VGPR layouts per CDNA5 ISA (CDNA5 ISA):
//   A 16x32 bf16: lane = row (lane&15); half=lane>>4; elems 0..7 -> K = 8*half..+7,
//                 elems 8..15 -> K = 16+8*half..+7  (two contiguous 16B chunks)
//   B 32x16 bf16: lane&15 = column N; same K order as A (two contiguous 16B chunks)
//   C/D f32 16x16: vgpr v, lane: M = v + 8*half, N = lane&15
__device__ __forceinline__ void gemm_tile_bf16(
    const bf16_t* __restrict__ A, int lda,
    const bf16_t* __restrict__ BT, int ldk,
    int mbase, int nbase, int K, int lane, v8f& acc)
{
  const int hf  = lane >> 4;
  const int l16 = lane & 15;
  const bf16_t* arow = A  + (size_t)(mbase + l16) * lda;
  const bf16_t* brow = BT + (size_t)(nbase + l16) * ldk;
#pragma unroll 4
  for (int k = 0; k < K; k += 32) {
    BF16x16 a, b;
    a.h[0] = *reinterpret_cast<const v8bf*>(arow + k + 8 * hf);
    a.h[1] = *reinterpret_cast<const v8bf*>(arow + k + 16 + 8 * hf);
    b.h[0] = *reinterpret_cast<const v8bf*>(brow + k + 8 * hf);
    b.h[1] = *reinterpret_cast<const v8bf*>(brow + k + 16 + 8 * hf);
    acc = __builtin_amdgcn_wmma_f32_16x16x32_bf16(false, a.v, false, b.v,
                                                  (short)0, acc, false, false);
  }
}

// ---------------- conversion helpers ----------------

__global__ void __launch_bounds__(256)
f32_to_bf16_kernel(const float* __restrict__ in, bf16_t* __restrict__ out, int n)
{
  int i = blockIdx.x * 256 + threadIdx.x;
  if (i < n) out[i] = (bf16_t)in[i];
}

// in: [K,N] row-major f32  ->  out: [N,K] row-major bf16 (i.e. B^T for WMMA feed)
__global__ void __launch_bounds__(256)
transpose_f32_to_bf16_kernel(const float* __restrict__ in, bf16_t* __restrict__ out,
                             int K, int N)
{
  int i = blockIdx.x * 256 + threadIdx.x;
  if (i < K * N) {
    int n = i / K;
    int k = i - n * K;
    out[(size_t)n * K + k] = (bf16_t)in[(size_t)k * N + n];
  }
}

// ---------------- phase 1: x_proj = inputs @ W_xh ----------------
// A = inputs bf16 [B*T, I], BT = WxhT [H, I], out xp bf16 [B*T, H].
// 2048 M-tiles x 64 N-tiles = 131072 waves = 16384 blocks of 256.
__global__ void __launch_bounds__(256)
xproj_kernel(const bf16_t* __restrict__ Abf, const bf16_t* __restrict__ WxhT,
             bf16_t* __restrict__ xp)
{
  const int lane = threadIdx.x & 31;
  const int w    = (blockIdx.x * 256 + threadIdx.x) >> 5;
  const int hf   = lane >> 4;
  const int l16  = lane & 15;
  const int mt = w >> 6;
  const int nt = w & 63;
  v8f acc = {};
  gemm_tile_bf16(Abf, II, WxhT, II, mt * 16, nt * 16, II, lane, acc);
  const int n = nt * 16 + l16;
#pragma unroll
  for (int v = 0; v < 8; ++v) {
    int m = mt * 16 + v + 8 * hf;
    xp[(size_t)m * HH + n] = (bf16_t)acc[v];
  }
}

// ---------------- phase 2: one launch per timestep ----------------
// Launch t = 0..TT (inclusive). Waves 0..255: h_t = tanh(xp_t + h_{t-1}@W_hh + b_h)
// (skipped at t==TT). Waves 256..383: out_{t-1} = h_{t-1} @ W_hy + b_y (skipped at
// t==0). Both consume only h_{t-1}, written by the previous launch -> stream order
// provides the dependency, no intra-kernel sync needed.
__global__ void __launch_bounds__(256)
rnn_step_kernel(const bf16_t* __restrict__ hprev, bf16_t* __restrict__ hcur,
                const bf16_t* __restrict__ xp, const bf16_t* __restrict__ WhhT,
                const bf16_t* __restrict__ WhyT, const float* __restrict__ bh,
                const float* __restrict__ by, float* __restrict__ out,
                float* __restrict__ finalh, int t)
{
  const int lane = threadIdx.x & 31;
  const int w    = (blockIdx.x * 256 + threadIdx.x) >> 5;
  const int hf   = lane >> 4;
  const int l16  = lane & 15;

  if (w < 256) {                      // hidden-state update
    if (t >= TT) return;
    const int mt = w >> 6;            // 4 M-tiles
    const int nt = w & 63;            // 64 N-tiles (H=1024)
    v8f acc = {};
    gemm_tile_bf16(hprev, HH, WhhT, HH, mt * 16, nt * 16, HH, lane, acc);
    const int n = nt * 16 + l16;
    const float bias = bh[n];
#pragma unroll
    for (int v = 0; v < 8; ++v) {
      int m = mt * 16 + v + 8 * hf;   // batch row
      float val = acc[v] + (float)xp[(size_t)(m * TT + t) * HH + n] + bias;
      val = tanhf(val);
      hcur[(size_t)m * HH + n] = (bf16_t)val;
      if (t == TT - 1) finalh[(size_t)m * HH + n] = val;
    }
  } else {                            // output projection of previous step
    if (t == 0) return;
    const int w2 = w - 256;
    const int mt = w2 >> 5;           // 4 M-tiles
    const int nt = w2 & 31;           // 32 N-tiles (O=512)
    v8f acc = {};
    gemm_tile_bf16(hprev, HH, WhyT, HH, mt * 16, nt * 16, HH, lane, acc);
    const int n = nt * 16 + l16;
    const float bias = by[n];
#pragma unroll
    for (int v = 0; v < 8; ++v) {
      int m = mt * 16 + v + 8 * hf;
      out[(size_t)m * TT * OO + (size_t)(t - 1) * OO + n] = acc[v] + bias;
    }
  }
}

// ---------------- host ----------------

extern "C" void kernel_launch(void* const* d_in, const int* in_sizes, int n_in,
                              void* d_out, int out_size, void* d_ws, size_t ws_size,
                              hipStream_t stream)
{
  const float* inputs = (const float*)d_in[0];  // [B,T,I]
  const float* h0     = (const float*)d_in[1];  // [B,H]
  const float* Wxh    = (const float*)d_in[2];  // [I,H]
  const float* Whh    = (const float*)d_in[3];  // [H,H]
  const float* Why    = (const float*)d_in[4];  // [H,O]
  const float* bh     = (const float*)d_in[5];  // [H]
  const float* by     = (const float*)d_in[6];  // [O]
  float* out    = (float*)d_out;                        // [B,T,O] then [B,H]
  float* finalh = out + (size_t)BB * TT * OO;

  char* ws = (char*)d_ws;
  // workspace layout (all offsets 128B-aligned); total ~105 MB
  bf16_t* inbf = (bf16_t*)(ws);                               // 33,554,432 B
  bf16_t* xp   = (bf16_t*)(ws + 33554432ull);                 // 67,108,864 B
  bf16_t* WxhT = (bf16_t*)(ws + 100663296ull);                //  1,048,576 B
  bf16_t* WhhT = (bf16_t*)(ws + 101711872ull);                //  2,097,152 B
  bf16_t* WhyT = (bf16_t*)(ws + 103809024ull);                //  1,048,576 B
  bf16_t* hb0  = (bf16_t*)(ws + 104857600ull);                //    131,072 B
  bf16_t* hb1  = (bf16_t*)(ws + 104988672ull);                //    131,072 B
  bf16_t* hb[2] = {hb0, hb1};

  // convert / transpose to bf16 WMMA-friendly layouts
  f32_to_bf16_kernel<<<(BB * TT * II + 255) / 256, 256, 0, stream>>>(
      inputs, inbf, BB * TT * II);
  transpose_f32_to_bf16_kernel<<<(II * HH + 255) / 256, 256, 0, stream>>>(
      Wxh, WxhT, II, HH);   // -> [H, I]
  transpose_f32_to_bf16_kernel<<<(HH * HH + 255) / 256, 256, 0, stream>>>(
      Whh, WhhT, HH, HH);   // -> [H, H]
  transpose_f32_to_bf16_kernel<<<(HH * OO + 255) / 256, 256, 0, stream>>>(
      Why, WhyT, HH, OO);   // -> [O, H]
  f32_to_bf16_kernel<<<(BB * HH + 255) / 256, 256, 0, stream>>>(h0, hb0, BB * HH);

  // big parallel GEMM: x_proj for all timesteps
  xproj_kernel<<<16384, 256, 0, stream>>>(inbf, WxhT, xp);

  // sequential scan; step t also emits out_{t-1}; t==TT emits only out_{TT-1}
  for (int t = 0; t <= TT; ++t) {
    rnn_step_kernel<<<48, 256, 0, stream>>>(
        hb[t & 1], hb[(t + 1) & 1], xp, WhhT, WhyT, bh, by, out, finalh, t);
  }
}
